// MossVLTextCrossAttention_36936718746208
// MI455X (gfx1250) — matrix operand order repaired and checked by reading, not compiled, not run on hardware
//
#include <hip/hip_runtime.h>
#include <math.h>

// MossVL text cross-attention for MI455X (gfx1250, wave32, WMMA).
// Preamble: convert hidden/cross to f16; convert+transpose Wq/Wk/Wv/Wo to f16.
// Then: [GEMM Q][GEMM K][GEMM V(transposed store)] -> [rmsnorm+mrope Q,K]
//       -> [flash GQA attention] -> [GEMM O -> f32 out].
// All matrix math: v_wmma_f32_16x16x32_f16. All LDS tile staging uses the
// CDNA5 async path: global_load_async_to_lds_b128 (+ s_wait_asynccnt),
// double-buffered so copies for tile kt+1 overlap the WMMAs on tile kt.

typedef __attribute__((ext_vector_type(16))) _Float16 v16h;
typedef __attribute__((ext_vector_type(8)))  float    v8f;

#define H_   16
#define KVH_ 4
#define D_   128
#define HID_ 2048
#define T_   2048
#define S_   4096

__device__ __forceinline__ v8f wmma32(const v16h& a, const v16h& b, const v8f& c) {
  return __builtin_amdgcn_wmma_f32_16x16x32_f16(false, a, false, b, (short)0, c,
                                                false, false);
}
__device__ __forceinline__ v8f zero8() {
  v8f z = {0.f, 0.f, 0.f, 0.f, 0.f, 0.f, 0.f, 0.f};
  return z;
}

// LDS byte offset of a generic pointer into shared memory (addrspace(3) cast,
// same idiom as HIP's __to_local).
__device__ __forceinline__ unsigned lds_u32(const _Float16* p) {
  return (unsigned)(unsigned long long)(__attribute__((address_space(3))) const _Float16*)p;
}
// CDNA5 async memory->LDS copy, 16B per lane, tracked by ASYNCcnt.
__device__ __forceinline__ void async_ld_b128(unsigned lds, const _Float16* gsrc) {
  asm volatile("global_load_async_to_lds_b128 %0, %1, off"
               :: "v"(lds), "v"((unsigned long long)gsrc) : "memory");
}
__device__ __forceinline__ void wait_async0() {
  asm volatile("s_wait_asynccnt 0x0" ::: "memory");
}

// ---------------------------------------------------------------------------
// Preamble converters.
// ---------------------------------------------------------------------------
__global__ void __launch_bounds__(256)
cvt_f32_f16(const float* __restrict__ in, _Float16* __restrict__ out, long n) {
  long i = ((long)blockIdx.x * 256 + threadIdx.x) * 8;
  if (i < n) {
#pragma unroll
    for (int j = 0; j < 8; ++j) out[i + j] = (_Float16)in[i + j];
  }
}

// in (K,N) f32 row-major  ->  out (N,K) f16 row-major (i.e. B^T)
__global__ void __launch_bounds__(256)
transpose_cvt(const float* __restrict__ in, _Float16* __restrict__ out,
              int K, int N) {
  __shared__ _Float16 tile[32][34];
  int k0 = blockIdx.y * 32, n0 = blockIdx.x * 32;
  int tx = threadIdx.x, ty = threadIdx.y;   // block (32, 8)
#pragma unroll
  for (int i = 0; i < 32; i += 8)
    tile[ty + i][tx] = (_Float16)in[(size_t)(k0 + ty + i) * N + n0 + tx];
  __syncthreads();
#pragma unroll
  for (int i = 0; i < 32; i += 8)
    out[(size_t)(n0 + ty + i) * K + k0 + tx] = tile[tx][ty + i];
}

// ---------------------------------------------------------------------------
// Tiled GEMM: C[M,N] = A[M,K] * B[K,N] with A f16 row-major and Bt = B^T f16
// row-major (N x K). 256 threads = 8 waves; block tile 128x128, BK=32;
// double-buffered LDS filled with global_load_async_to_lds_b128.
// storeMode: 0 = f16 row-major C, 1 = f16 transposed C (C^T, N x M), 2 = f32.
// ---------------------------------------------------------------------------
#define BM 128
#define BN 128
#define BK 32
#define LDT 40   // halves per staged row (80B pitch, 16B aligned, no conflicts)

__global__ void __launch_bounds__(256)
gemm_f16(const _Float16* __restrict__ A, const _Float16* __restrict__ Bt,
         void* __restrict__ Cout, int M, int N, int K, int storeMode) {
  __shared__ _Float16 sA[2][BM * LDT];
  __shared__ _Float16 sB[2][BN * LDT];

  const int tid  = threadIdx.x;
  const int lane = tid & 31;
  const int wave = tid >> 5;      // 0..7
  const int wm   = wave >> 2;     // 0..1 : 64 rows each
  const int wn   = wave & 3;      // 0..3 : 32 cols each
  const int half = lane >> 4;
  const int l16  = lane & 15;

  const int blockM = blockIdx.y * BM;
  const int blockN = blockIdx.x * BN;

  // staging role: each thread owns a 32B (2x16B) run of one 64B tile row
  const int srow = tid >> 1;            // 0..127 (tile row, both A and Bt)
  const int soff = (tid & 1) * 16;      // halves offset: 0 or 16

  v8f acc[4][2];
#pragma unroll
  for (int mt = 0; mt < 4; ++mt)
#pragma unroll
    for (int nt = 0; nt < 2; ++nt) acc[mt][nt] = zero8();

  const int kTiles = K / BK;
  const _Float16* aRow = A  + (size_t)(blockM + srow) * K + soff;
  const _Float16* bRow = Bt + (size_t)(blockN + srow) * K + soff;

  {   // prologue: async-stage tile 0 into buffer 0
    unsigned la = lds_u32(&sA[0][srow * LDT + soff]);
    unsigned lb = lds_u32(&sB[0][srow * LDT + soff]);
    async_ld_b128(la,      aRow);
    async_ld_b128(la + 16, aRow + 8);
    async_ld_b128(lb,      bRow);
    async_ld_b128(lb + 16, bRow + 8);
    wait_async0();
  }
  __syncthreads();

  for (int kt = 0; kt < kTiles; ++kt) {
    const int cur = kt & 1;
    const bool more = (kt + 1) < kTiles;

    if (more) {  // async-stage tile kt+1 into the other buffer
      unsigned la = lds_u32(&sA[cur ^ 1][srow * LDT + soff]);
      unsigned lb = lds_u32(&sB[cur ^ 1][srow * LDT + soff]);
      const _Float16* ap = aRow + (size_t)(kt + 1) * BK;
      const _Float16* bp = bRow + (size_t)(kt + 1) * BK;
      async_ld_b128(la,      ap);
      async_ld_b128(la + 16, ap + 8);
      async_ld_b128(lb,      bp);
      async_ld_b128(lb + 16, bp + 8);
      if (kt + 2 < kTiles) {  // pull tile kt+2 toward L2
        __builtin_prefetch(aRow + (size_t)(kt + 2) * BK, 0, 1);
        __builtin_prefetch(bRow + (size_t)(kt + 2) * BK, 0, 1);
      }
    }

    // A fragments (16x32): k = half*16 + e  -> 2x ds_load_b128
    v16h afrag[4];
#pragma unroll
    for (int mt = 0; mt < 4; ++mt) {
      int row = wm * 64 + mt * 16 + l16;
#pragma unroll
      for (int e = 0; e < 16; ++e) {
        int k = half * 16 + e;
        afrag[mt][e] = sA[cur][row * LDT + k];
      }
    }
    // B fragments (32x16): k = half*16 + e  -> 2x ds_load_b128
    v16h bfrag[2];
#pragma unroll
    for (int nt = 0; nt < 2; ++nt) {
      int col = wn * 32 + nt * 16 + l16;
#pragma unroll
      for (int e = 0; e < 16; ++e)
        bfrag[nt][e] = sB[cur][col * LDT + half * 16 + e];
    }
#pragma unroll
    for (int mt = 0; mt < 4; ++mt)
#pragma unroll
      for (int nt = 0; nt < 2; ++nt)
        acc[mt][nt] = wmma32(afrag[mt], bfrag[nt], acc[mt][nt]);

    if (more) wait_async0();
    __syncthreads();
  }

  // C/D layout: row = v + 8*half, col = lane&15
#pragma unroll
  for (int mt = 0; mt < 4; ++mt)
#pragma unroll
    for (int nt = 0; nt < 2; ++nt) {
      int col = blockN + wn * 32 + nt * 16 + l16;
      int rbase = blockM + wm * 64 + mt * 16 + half * 8;
      if (storeMode == 0) {
        _Float16* C16 = (_Float16*)Cout;
#pragma unroll
        for (int v = 0; v < 8; ++v)
          C16[(size_t)(rbase + v) * N + col] = (_Float16)acc[mt][nt][v];
      } else if (storeMode == 1) {  // C^T: 8 contiguous halves -> b128 store
        _Float16* C16 = (_Float16*)Cout;
#pragma unroll
        for (int v = 0; v < 8; ++v)
          C16[(size_t)col * M + rbase + v] = (_Float16)acc[mt][nt][v];
      } else {
        float* C32 = (float*)Cout;
#pragma unroll
        for (int v = 0; v < 8; ++v)
          C32[(size_t)(rbase + v) * N + col] = acc[mt][nt][v];
      }
    }
}

// ---------------------------------------------------------------------------
// RMSNorm + mRoPE (neox), in place on f16, f32 math. One wave per (row, head).
// ---------------------------------------------------------------------------
__global__ void __launch_bounds__(32)
rmsnorm_rope(_Float16* __restrict__ x, const float* __restrict__ w,
             const int* __restrict__ pos, int nheads, int nrows) {
  const int idx  = blockIdx.x;        // row * nheads + head
  const int row  = idx / nheads;
  const int lane = threadIdx.x;
  _Float16* p = x + (size_t)idx * D_;

  float x1a = (float)p[lane],      x1b = (float)p[lane + 32];
  float x2a = (float)p[lane + 64], x2b = (float)p[lane + 96];
  float ss = x1a * x1a + x1b * x1b + x2a * x2a + x2b * x2b;
#pragma unroll
  for (int m = 16; m >= 1; m >>= 1) ss += __shfl_xor(ss, m, 32);
  const float r = rsqrtf(ss * (1.0f / 128.0f) + 1e-6f);
  x1a *= r * w[lane];      x1b *= r * w[lane + 32];
  x2a *= r * w[lane + 64]; x2b *= r * w[lane + 96];

  const float LOG_THETA = 13.815510557964274f;  // ln(1e6)
  int seca = (lane < 16) ? 0 : 1;
  int pb   = lane + 32;
  int secb = (pb < 40) ? 1 : 2;
  float inva = __expf(-(2.0f * lane / 128.0f) * LOG_THETA);
  float invb = __expf(-(2.0f * pb   / 128.0f) * LOG_THETA);
  float fa = (float)pos[(size_t)seca * nrows + row] * inva;
  float fb = (float)pos[(size_t)secb * nrows + row] * invb;
  float ca = __cosf(fa), sa = __sinf(fa);
  float cb = __cosf(fb), sb = __sinf(fb);

  p[lane]      = (_Float16)(x1a * ca - x2a * sa);
  p[lane + 64] = (_Float16)(x2a * ca + x1a * sa);
  p[lane + 32] = (_Float16)(x1b * cb - x2b * sb);
  p[lane + 96] = (_Float16)(x2b * cb + x1b * sb);
}

// ---------------------------------------------------------------------------
// Flash GQA attention, all-f16 operands. Block = 128 threads = 4 waves;
// grid (T/64, H). K tile (32x128) staged row-major; V comes pre-transposed
// (vT[d][s]) so its tile is also a plain contiguous copy. Both are staged
// with async b128 copies, double-buffered.
// ---------------------------------------------------------------------------
#define SKLD  (D_ + 8)   // 272B pitch
#define SVLD  40         // 80B pitch
#define SPLD  40

__global__ void __launch_bounds__(128)
attention(const _Float16* __restrict__ q, const _Float16* __restrict__ k,
          const _Float16* __restrict__ vT, _Float16* __restrict__ o) {
  __shared__ _Float16 sK[2][32 * SKLD];
  __shared__ _Float16 sVt[2][D_ * SVLD];
  __shared__ _Float16 sP[4][16 * SPLD];

  const int tid   = threadIdx.x;
  const int lane  = tid & 31;
  const int wave  = tid >> 5;
  const int half  = lane >> 4;
  const int l16   = lane & 15;
  const int h     = blockIdx.y;
  const int kvh   = h >> 2;                  // G = H/KV = 4
  const int qbase = blockIdx.x * 64 + wave * 16;

  // staging roles
  const int krow = tid >> 2, kseg = (tid & 3) * 64;   // sK: 4x16B per thread
  const int vrow = tid;                               // sVt: one 64B row each
  const _Float16* kSrc = k + ((size_t)krow * KVH_ + kvh) * D_ + kseg / 2;
  const _Float16* vSrc = vT + ((size_t)(kvh * D_ + vrow) * S_);

  // Q A-fragments over 4 chunks of D; 1/sqrt(D) folded in.
  v16h qf[4];
  {
    const _Float16* qp = q + ((size_t)(qbase + l16) * H_ + h) * D_;
#pragma unroll
    for (int c = 0; c < 4; ++c)
#pragma unroll
      for (int e = 0; e < 16; ++e) {
        int kk = c * 32 + half * 16 + e;
        qf[c][e] = (_Float16)((float)qp[kk] * 0.08838834764831845f);
      }
  }

  v8f oacc[8];
  float run_max[8], run_sum[8];
#pragma unroll
  for (int i = 0; i < 8; ++i) { oacc[i] = zero8(); run_max[i] = -1e30f; run_sum[i] = 0.f; }

  {   // prologue: stage s-tile 0 into buffer 0
    unsigned lk = lds_u32(&sK[0][krow * SKLD]) + kseg;
    unsigned lv = lds_u32(&sVt[0][vrow * SVLD]);
#pragma unroll
    for (int i = 0; i < 4; ++i) async_ld_b128(lk + i * 16, kSrc + i * 8);
#pragma unroll
    for (int i = 0; i < 4; ++i) async_ld_b128(lv + i * 16, vSrc + i * 8);
    wait_async0();
  }
  __syncthreads();

  const int sTiles = S_ / 32;
  for (int st = 0; st < sTiles; ++st) {
    const int cur = st & 1;
    const bool more = (st + 1) < sTiles;
    if (more) {  // async-stage tile st+1
      unsigned lk = lds_u32(&sK[cur ^ 1][krow * SKLD]) + kseg;
      unsigned lv = lds_u32(&sVt[cur ^ 1][vrow * SVLD]);
      const _Float16* kp = kSrc + (size_t)(st + 1) * 32 * KVH_ * D_;
      const _Float16* vp = vSrc + (size_t)(st + 1) * 32;
#pragma unroll
      for (int i = 0; i < 4; ++i) async_ld_b128(lk + i * 16, kp + i * 8);
#pragma unroll
      for (int i = 0; i < 4; ++i) async_ld_b128(lv + i * 16, vp + i * 8);
    }

    // scores: two 16x16 tiles (s columns), K-dim = D in 4 chunks of 32
    v8f sc[2] = { zero8(), zero8() };
#pragma unroll
    for (int nt = 0; nt < 2; ++nt)
#pragma unroll
      for (int c = 0; c < 4; ++c) {
        v16h bf;  // bf[e] = K[s_col][d], contiguous in d
#pragma unroll
        for (int e = 0; e < 16; ++e)
          bf[e] = sK[cur][(nt * 16 + l16) * SKLD + c * 32 + half * 16 + e];
        sc[nt] = wmma32(qf[c], bf, sc[nt]);
      }

    // online softmax; row m = v + 8*half; width-16 reductions
    float corr[8];
#pragma unroll
    for (int vv = 0; vv < 8; ++vv) {
      float mx = fmaxf(sc[0][vv], sc[1][vv]);
#pragma unroll
      for (int msk = 8; msk >= 1; msk >>= 1) mx = fmaxf(mx, __shfl_xor(mx, msk, 32));
      float nm = fmaxf(run_max[vv], mx);
      corr[vv] = __expf(run_max[vv] - nm);
      run_max[vv] = nm;
      float p0 = __expf(sc[0][vv] - nm);
      float p1 = __expf(sc[1][vv] - nm);
      sc[0][vv] = p0; sc[1][vv] = p1;
      float rs = p0 + p1;
#pragma unroll
      for (int msk = 8; msk >= 1; msk >>= 1) rs += __shfl_xor(rs, msk, 32);
      run_sum[vv] = run_sum[vv] * corr[vv] + rs;
    }
#pragma unroll
    for (int nt2 = 0; nt2 < 8; ++nt2)
#pragma unroll
      for (int vv = 0; vv < 8; ++vv) oacc[nt2][vv] *= corr[vv];

    // P (C-layout) -> LDS -> A-fragment layout
    _Float16* sp = &sP[wave][0];
#pragma unroll
    for (int vv = 0; vv < 8; ++vv) {
      int m = vv + 8 * half;
      sp[m * SPLD + l16]      = (_Float16)sc[0][vv];
      sp[m * SPLD + 16 + l16] = (_Float16)sc[1][vv];
    }
    __syncthreads();
    v16h pf;
#pragma unroll
    for (int e = 0; e < 16; ++e) {
      int kk = half * 16 + e;
      pf[e] = sp[l16 * SPLD + kk];
    }
    // oacc += P(16x32) @ V(32x128); V fragment contiguous in the s direction
#pragma unroll
    for (int nt2 = 0; nt2 < 8; ++nt2) {
      v16h bf;
#pragma unroll
      for (int e = 0; e < 16; ++e)
        bf[e] = sVt[cur][(nt2 * 16 + l16) * SVLD + half * 16 + e];
      oacc[nt2] = wmma32(pf, bf, oacc[nt2]);
    }

    if (more) wait_async0();
    __syncthreads();
  }

  // normalize and store f16: o[t, h*D + d]
#pragma unroll
  for (int nt2 = 0; nt2 < 8; ++nt2) {
    int d = nt2 * 16 + l16;
#pragma unroll
    for (int vv = 0; vv < 8; ++vv) {
      int t = qbase + vv + 8 * half;
      o[(size_t)t * (H_ * D_) + h * D_ + d] = (_Float16)(oacc[nt2][vv] / run_sum[vv]);
    }
  }
}

// ---------------------------------------------------------------------------
extern "C" void kernel_launch(void* const* d_in, const int* in_sizes, int n_in,
                              void* d_out, int out_size, void* d_ws, size_t ws_size,
                              hipStream_t stream) {
  (void)in_sizes; (void)n_in; (void)out_size; (void)ws_size;

  const float* hidden = (const float*)d_in[0];   // (T, HID)
  const float* cross  = (const float*)d_in[1];   // (S, HID)
  const float* Wq     = (const float*)d_in[2];   // (HID, H*D)
  const float* Wk     = (const float*)d_in[3];   // (HID, KV*D)
  const float* Wv     = (const float*)d_in[4];   // (HID, KV*D)
  const float* Wo     = (const float*)d_in[5];   // (H*D, HID)
  const float* qnw    = (const float*)d_in[6];
  const float* knw    = (const float*)d_in[7];
  const int*   pos    = (const int*)d_in[8];     // (3, T)
  const int*   vpos   = (const int*)d_in[9];     // (3, S)
  float* out = (float*)d_out;                    // (T, HID)

  _Float16* h16  = (_Float16*)d_ws;                        // T*HID
  _Float16* c16  = h16  + (size_t)T_ * HID_;               // S*HID
  _Float16* WqT  = c16  + (size_t)S_ * HID_;               // (H*D)  x HID
  _Float16* WkT  = WqT  + (size_t)(H_ * D_) * HID_;        // (KV*D) x HID
  _Float16* WvT  = WkT  + (size_t)(KVH_ * D_) * HID_;      // (KV*D) x HID
  _Float16* WoT  = WvT  + (size_t)(KVH_ * D_) * HID_;      // HID    x (H*D)
  _Float16* qbuf = WoT  + (size_t)HID_ * (H_ * D_);        // T x (H*D)
  _Float16* kbuf = qbuf + (size_t)T_ * H_ * D_;            // S x (KV*D)
  _Float16* vT   = kbuf + (size_t)S_ * KVH_ * D_;          // (KV*D) x S
  _Float16* abuf = vT   + (size_t)(KVH_ * D_) * S_;        // T x (H*D)

  // Preamble: f16 conversion (+ weight transposes)
  cvt_f32_f16<<<(T_ * HID_) / (256 * 8), 256, 0, stream>>>(hidden, h16, (long)T_ * HID_);
  cvt_f32_f16<<<(S_ * HID_) / (256 * 8), 256, 0, stream>>>(cross, c16, (long)S_ * HID_);
  transpose_cvt<<<dim3((H_ * D_) / 32, HID_ / 32), dim3(32, 8), 0, stream>>>(Wq, WqT, HID_, H_ * D_);
  transpose_cvt<<<dim3((KVH_ * D_) / 32, HID_ / 32), dim3(32, 8), 0, stream>>>(Wk, WkT, HID_, KVH_ * D_);
  transpose_cvt<<<dim3((KVH_ * D_) / 32, HID_ / 32), dim3(32, 8), 0, stream>>>(Wv, WvT, HID_, KVH_ * D_);
  transpose_cvt<<<dim3(HID_ / 32, (H_ * D_) / 32), dim3(32, 8), 0, stream>>>(Wo, WoT, H_ * D_, HID_);

  // Projections
  gemm_f16<<<dim3((H_ * D_) / BN, T_ / BM), 256, 0, stream>>>(h16, WqT, qbuf, T_, H_ * D_, HID_, 0);
  gemm_f16<<<dim3((KVH_ * D_) / BN, S_ / BM), 256, 0, stream>>>(c16, WkT, kbuf, S_, KVH_ * D_, HID_, 0);
  gemm_f16<<<dim3((KVH_ * D_) / BN, S_ / BM), 256, 0, stream>>>(c16, WvT, vT, S_, KVH_ * D_, HID_, 1);

  // RMSNorm + mRoPE on Q and K (in place, f16)
  rmsnorm_rope<<<T_ * H_, 32, 0, stream>>>(qbuf, qnw, pos, H_, T_);
  rmsnorm_rope<<<S_ * KVH_, 32, 0, stream>>>(kbuf, knw, vpos, KVH_, S_);

  // Flash GQA attention
  attention<<<dim3(T_ / 64, H_), 128, 0, stream>>>(qbuf, kbuf, vT, abuf);

  // Output projection -> f32
  gemm_f16<<<dim3(HID_ / BN, T_ / BM), 256, 0, stream>>>(abuf, WoT, out, T_, HID_, H_ * D_, 2);
}
